// ECGCLayer_51505247814192
// MI455X (gfx1250) — matrix operand order repaired and checked by reading, not compiled, not run on hardware
//
#include <hip/hip_runtime.h>
#include <hip/hip_bf16.h>

#define BB 2
#define NN 512
#define DD 128
#define EE 64
#define LN_EPS 1e-5f

typedef __attribute__((ext_vector_type(16))) _Float16 v16h;
typedef __attribute__((ext_vector_type(8)))  float    v8f;

__device__ __forceinline__ float fast_sigmoid(float x) {
    // v_mul + v_exp_f32 + v_add + v_rcp_f32 (fast rcp, ~1 ulp)
    return __builtin_amdgcn_rcpf(1.0f + __expf(-x));
}

// ---------------------------------------------------------------------------
// Kernel 1: hW1a[bi][e] = b1[e] + sum_d h[bi][d] * W1[d][e]     (bi = b*N+i)
// ---------------------------------------------------------------------------
__global__ void k_prep_hw1a(const float* __restrict__ H,
                            const float* __restrict__ W1,
                            const float* __restrict__ B1,
                            float* __restrict__ HW1A) {
    __shared__ float s_h[DD];
    const int bi = blockIdx.x;
    const int t  = threadIdx.x;            // 0..63  (t == e)
    s_h[t]      = H[bi * DD + t];
    s_h[t + 64] = H[bi * DD + t + 64];
    __syncthreads();
    float acc = B1[t];
    #pragma unroll 8
    for (int d = 0; d < DD; ++d)
        acc = fmaf(s_h[d], W1[d * EE + t], acc);
    HW1A[bi * EE + t] = acc;
}

// ---------------------------------------------------------------------------
// Kernel 2: repack W2 (64x128 f32, row major) into f16 WMMA B-operand layout.
// Index = ((ks*8 + dtile)*8 + v)*32 + lane ; each uint holds K={k,k+1} halves.
//   lane<16 : n = dtile*16+lane,    k = ks*32 + 2v
//   lane>=16: n = dtile*16+lane-16, k = ks*32 + 16 + 2v
// ---------------------------------------------------------------------------
__global__ void k_pack_w2(const float* __restrict__ W2,
                          unsigned int* __restrict__ W2P) {
    const int tid = blockIdx.x * blockDim.x + threadIdx.x;   // 0..4095
    if (tid >= 2 * 8 * 8 * 32) return;
    const int lane = tid & 31;
    const int v    = (tid >> 5) & 7;
    const int dt   = (tid >> 8) & 7;
    const int ks   = tid >> 11;
    const int n    = dt * 16 + (lane & 15);
    const int k    = ks * 32 + ((lane >> 4) * 16) + 2 * v;
    union { _Float16 hh[2]; unsigned int u; } pk;
    pk.hh[0] = (_Float16)W2[k * DD + n];
    pk.hh[1] = (_Float16)W2[(k + 1) * DD + n];
    W2P[tid] = pk.u;
}

// ---------------------------------------------------------------------------
// Kernel 3 (main, WMMA): one WG of 256 threads (8 wave32) per (b,i).
// agg[bi][d] = h[bi][d] * sum_j mask[bi][j]*sigmoid( (hid_j @ W2)[d] + b2[d] )
// hid_j[e] = relu( hW1a[bi][e] + sum_c ef[bi][j][c]*W1[D+c][e] )
// LDS hid tile is double-buffered -> single barrier per j-tile.
// ---------------------------------------------------------------------------
__global__ void __launch_bounds__(256)
k_main_gates(const float* __restrict__ H,
             const float* __restrict__ EF,          // (B,N,N,3)
             const unsigned char* __restrict__ MSK, // (B,N,N) bool bytes
             const float* __restrict__ W1,          // (D+3, E) ; rows D..D+2
             const float* __restrict__ B2,          // (D,)
             const float* __restrict__ HW1A,        // (B*N, E)
             const unsigned int* __restrict__ W2P,  // packed f16 B tiles
             float* __restrict__ AGG) {             // (B*N, D)
    constexpr int ROWB = 144;                       // 16 pad bytes per row
    constexpr int BUFB = 16 * ROWB;                 // 2304 B per buffer
    __shared__ __align__(16) unsigned char s_hid[2 * BUFB];

    const int bi = blockIdx.x;                      // b*N + i
    const int t  = threadIdx.x;
    const int l  = t & 31;                          // lane within wave32
    const int w  = t >> 5;                          // wave id = d-tile (0..7)

    // ---- staging-role constants (per thread): j row + 4 consecutive e ----
    const int jloc = t & 15;
    const int e0   = (t >> 4) * 4;
    float hb[4], wa[4], wb[4], wc[4];
    #pragma unroll
    for (int q = 0; q < 4; ++q) {
        hb[q] = HW1A[bi * EE + e0 + q];
        wa[q] = W1[(DD + 0) * EE + e0 + q];
        wb[q] = W1[(DD + 1) * EE + e0 + q];
        wc[q] = W1[(DD + 2) * EE + e0 + q];
    }

    // ---- WMMA-role constants: B operands (W2 f16 tiles), b2 scalar ----
    union BU { unsigned int u[8]; v16h h; };
    BU b0, b1;
    #pragma unroll
    for (int v = 0; v < 8; ++v) {
        b0.u[v] = W2P[((0 * 8 + w) * 8 + v) * 32 + l];
        b1.u[v] = W2P[((1 * 8 + w) * 8 + v) * 32 + l];
    }
    const float b2v    = B2[w * 16 + (l & 15)];
    const int   rowoff = (l & 15) * ROWB;           // A row (j) for this lane
    const int   off0   = (l >> 4) * 32;             // lane-half byte offset
    const int   mbase  = (l >> 4) * 8;              // rows M handled per lane

    float colsum = 0.0f;

    for (int jt = 0; jt < NN / 16; ++jt) {
        const int j0 = jt * 16;
        unsigned char* buf = s_hid + (jt & 1) * BUFB;

        // -------- stage hid tile (16 j x 64 e, f16) into LDS --------
        {
            const long eoff = ((long)bi * NN + (j0 + jloc)) * 3;
            const float f0 = EF[eoff + 0];
            const float f1 = EF[eoff + 1];
            const float f2 = EF[eoff + 2];
            union { _Float16 hh[4]; uint2 u2; } pk;
            #pragma unroll
            for (int q = 0; q < 4; ++q) {
                float v = hb[q];
                v = fmaf(f0, wa[q], v);
                v = fmaf(f1, wb[q], v);
                v = fmaf(f2, wc[q], v);
                v = fmaxf(v, 0.0f);
                pk.hh[q] = (_Float16)v;
            }
            *(uint2*)(buf + jloc * ROWB + e0 * 2) = pk.u2;
        }
        __syncthreads();

        // -------- 2 x v_wmma_f32_16x16x32_f16 (K = 64) --------
        union AU { float4 q[2]; v16h h; };
        AU a0, a1;
        a0.q[0] = *(const float4*)(buf + rowoff + off0 + 0);
        a0.q[1] = *(const float4*)(buf + rowoff + off0 + 16);
        a1.q[0] = *(const float4*)(buf + rowoff + off0 + 64);
        a1.q[1] = *(const float4*)(buf + rowoff + off0 + 80);

        v8f c = {};
        c = __builtin_amdgcn_wmma_f32_16x16x32_f16(
                false, a0.h, false, b0.h, (short)0, c, false, false);
        c = __builtin_amdgcn_wmma_f32_16x16x32_f16(
                false, a1.h, false, b1.h, (short)0, c, false, false);

        // -------- sigmoid + mask + column-sum fold --------
        const unsigned char* mp = MSK + (long)bi * NN + j0 + mbase;
        #pragma unroll
        for (int r = 0; r < 8; ++r) {
            const float g = fast_sigmoid(c[r] + b2v);
            colsum = fmaf(g, (float)mp[r], colsum);
        }
        // no trailing barrier: next iteration writes the other buffer, and
        // the barrier inside it orders this buffer's reuse two tiles later.
    }

    // combine the two lane halves (rows M 0-7 and 8-15 of every tile)
    colsum += __shfl_xor(colsum, 16, 32);
    if (l < 16) {
        const int d = w * 16 + l;
        AGG[bi * DD + d] = H[bi * DD + d] * colsum;
    }
}

// ---------------------------------------------------------------------------
// Kernel 4: update MLP + residual + LayerNorm. 128 threads per (b,i).
// ---------------------------------------------------------------------------
__global__ void k_update_ln(const float* __restrict__ H,
                            const float* __restrict__ AGG,
                            const float* __restrict__ U1,
                            const float* __restrict__ BU1,
                            const float* __restrict__ U2,
                            const float* __restrict__ BU2,
                            const float* __restrict__ GAMMA,
                            const float* __restrict__ BETA,
                            float* __restrict__ OUT) {
    __shared__ float s_in[2 * DD];
    __shared__ float s_t1[DD];
    __shared__ float s_r1[DD];
    __shared__ float s_r2[DD];
    const int bi = blockIdx.x;
    const int d  = threadIdx.x;            // 0..127
    s_in[d]      = H[bi * DD + d];
    s_in[d + DD] = AGG[bi * DD + d];
    __syncthreads();

    float t1 = BU1[d];
    #pragma unroll 8
    for (int k = 0; k < 2 * DD; ++k)
        t1 = fmaf(s_in[k], U1[k * DD + d], t1);
    s_t1[d] = fmaxf(t1, 0.0f);
    __syncthreads();

    float upd = BU2[d];
    #pragma unroll 8
    for (int k = 0; k < DD; ++k)
        upd = fmaf(s_t1[k], U2[k * DD + d], upd);

    const float x = s_in[d] + upd;
    s_r1[d] = x;
    s_r2[d] = x * x;
    __syncthreads();
    for (int off = DD / 2; off > 0; off >>= 1) {
        if (d < off) { s_r1[d] += s_r1[d + off]; s_r2[d] += s_r2[d + off]; }
        __syncthreads();
    }
    const float mu  = s_r1[0] * (1.0f / DD);
    const float var = s_r2[0] * (1.0f / DD) - mu * mu;
    OUT[bi * DD + d] = GAMMA[d] * (x - mu) * rsqrtf(var + LN_EPS) + BETA[d];
}

// ---------------------------------------------------------------------------
extern "C" void kernel_launch(void* const* d_in, const int* in_sizes, int n_in,
                              void* d_out, int out_size, void* d_ws, size_t ws_size,
                              hipStream_t stream) {
    const float*         H    = (const float*)d_in[0];
    const float*         EF   = (const float*)d_in[1];
    const unsigned char* MSK  = (const unsigned char*)d_in[2]; // jax bool bytes
    const float*         W1   = (const float*)d_in[3];
    const float*         B1   = (const float*)d_in[4];
    const float*         W2   = (const float*)d_in[5];
    const float*         B2   = (const float*)d_in[6];
    const float*         U1   = (const float*)d_in[7];
    const float*         BU1  = (const float*)d_in[8];
    const float*         U2   = (const float*)d_in[9];
    const float*         BU2  = (const float*)d_in[10];
    const float*         GAM  = (const float*)d_in[11];
    const float*         BET  = (const float*)d_in[12];
    float*               OUT  = (float*)d_out;

    // workspace layout
    char* ws = (char*)d_ws;
    float*        AGG  = (float*)ws;                                 // 512 KB
    float*        HW1A = (float*)(ws + (size_t)BB * NN * DD * 4);    // 256 KB
    unsigned int* W2P  = (unsigned int*)(ws + (size_t)BB * NN * DD * 4
                                            + (size_t)BB * NN * EE * 4); // 16 KB

    const int NWG = BB * NN;   // 1024

    k_prep_hw1a<<<NWG, 64, 0, stream>>>(H, W1, B1, HW1A);
    k_pack_w2<<<16, 256, 0, stream>>>(W2, W2P);
    k_main_gates<<<NWG, 256, 0, stream>>>(H, EF, MSK, W1, B2, HW1A, W2P, AGG);
    k_update_ln<<<NWG, 128, 0, stream>>>(H, AGG, U1, BU1, U2, BU2, GAM, BET, OUT);
}
